// complex_feature_renet_72739566125508
// MI455X (gfx1250) — compile-verified, matrix operand order checked by reading, not run on hardware
//
#include <hip/hip_runtime.h>
#include <hip/hip_bf16.h>

// ---------------------------------------------------------------------------
// MI455X (gfx1250, wave32) fused conv/linear ResNet.
// Every layer is a GEMM on v_wmma_f32_16x16x32_f16 (f32 accumulate):
//   MODE 0: "lin"  — A row-major [M,K] (rows contiguous in NCHW)
//   MODE 1: "conv" — A rows im2col-gathered on the fly (kh x 1 conv along H);
//                    row decomposition hoisted out of the K loop.
// Block tile 64(M) x 64(N), K-step 32; 8 wave32s, each wave owns a 16x32
// output tile (2 WMMAs / K-step). Double-buffered LDS; K-tile kt+1 is staged
// into registers (branchless, clamped addresses, int32 offsets) while tile kt
// feeds the WMMAs; LDS writes are packed 4-half ds_store_b64. One barrier per
// K-step. fp16 intermediates staged through the 192MB L2.
// ---------------------------------------------------------------------------

typedef __attribute__((ext_vector_type(16))) _Float16 v16h;
typedef __attribute__((ext_vector_type(4)))  _Float16 v4h;
typedef __attribute__((ext_vector_type(8)))  float    v8f;

#define BM 64
#define BN 64
#define BK 32
#define LDS_STRIDE 48   // halves: 96B row stride -> 16-half frags 32B aligned

__device__ __forceinline__ int iclamp(int v, int lo, int hi) {
  return v < lo ? lo : (v > hi ? hi : v);
}

// ---------------------------- elementwise helpers ---------------------------

__global__ void cvt_f32_f16_kernel(const float* __restrict__ in,
                                   _Float16* __restrict__ out, int n) {
  int i = blockIdx.x * blockDim.x + threadIdx.x;
  if (i < n) out[i] = (_Float16)in[i];
}

__global__ void add_f16_kernel(const _Float16* __restrict__ a,
                               const _Float16* __restrict__ b,
                               _Float16* __restrict__ o, int n) {
  int i = blockIdx.x * blockDim.x + threadIdx.x;
  if (i < n) o[i] = (_Float16)((float)a[i] + (float)b[i]);
}

__global__ void add_f16_out_f32_kernel(const _Float16* __restrict__ a,
                                       const _Float16* __restrict__ b,
                                       float* __restrict__ o, int n) {
  int i = blockIdx.x * blockDim.x + threadIdx.x;
  if (i < n) o[i] = (float)a[i] + (float)b[i];
}

// ------------------------------- WMMA GEMM ---------------------------------
// D[M,N] = act( A[M,K] * Wt[N,K]^T + bias[N] )
template <int MODE, int KH, int STORE, int RELU>
__global__ __launch_bounds__(256)
void wmma_gemm_f16(const _Float16* __restrict__ A,
                   const _Float16* __restrict__ Wt,
                   const float* __restrict__ bias,
                   _Float16* __restrict__ out,
                   int M, int N, int K,
                   int C, int H, int W, int pad, int H_out) {
  __shared__ __align__(64) _Float16 As[2][BM][LDS_STRIDE];
  __shared__ __align__(64) _Float16 Bs[2][BN][LDS_STRIDE];

  const int tid   = threadIdx.x;
  const int m_blk = blockIdx.x * BM;
  const int n_blk = blockIdx.y * BN;

  const int wv   = tid >> 5;
  const int lane = tid & 31;
  const int wm   = (wv & 3) << 4;    // wave M sub-tile: 0/16/32/48
  const int wn   = (wv >> 2) << 5;   // wave N sub-tile: 0/32
  const int khi  = (lane >> 4) << 4; // lane's K-half of the fragment: 0 or 16

  // Staging map: each thread owns a 4-half k-quad on 2 rows (rr, rr+32) of
  // both tiles -> 4 packed ds_store_b64 per K-step.
  const int kq = (tid & 7) << 2;  // 0,4,...,28
  const int rr = tid >> 3;        // 0..31

  // ---- hoisted per-row gather state (computed once, not per K-step) ----
  int aoff[2];              // MODE 0: m*K element offset
  int abase[2];             // MODE 1: b*C*H*W + w
  int hrow[2];              // MODE 1: h of the output row
  const int HW_out = H_out * W;
#pragma unroll
  for (int j = 0; j < 2; ++j) {
    const int m = m_blk + rr + 32 * j;  // M % 64 == 0 always (B = 1024)
    if (MODE == 0) {
      aoff[j] = m * K;
    } else {
      const int b  = m / HW_out;
      const int hw = m - b * HW_out;
      const int h  = hw / W;
      const int w  = hw - h * W;
      abase[j] = b * (C * H * W) + w;
      hrow[j]  = h;
    }
  }
  int  woff[2];
  bool gok[2];
#pragma unroll
  for (int j = 0; j < 2; ++j) {
    const int gn = n_blk + rr + 32 * j;
    gok[j]  = gn < N;
    woff[j] = (gok[j] ? gn : 0) * K;
  }

  // ---- branchless register staging of one K-tile ----
  v4h ra[2], rb[2];
  auto stage_regs = [&](int kt) {
    const int kbase = kt * BK + kq;
#pragma unroll
    for (int q = 0; q < 4; ++q) {
      const int  k   = kbase + q;
      const bool kok = k < K;
      const int  kc  = kok ? k : 0;     // clamped: address stays in-bounds
      if (MODE == 0) {
#pragma unroll
        for (int j = 0; j < 2; ++j) {
          const _Float16 v = A[aoff[j] + kc];
          ra[j][q] = kok ? v : (_Float16)0.0f;   // v_cndmask, no exec churn
        }
      } else {
        const int i    = kc / KH;       // KH compile-time
        const int hoff = (kc - i * KH) - pad;
        const int ib   = i * H;
#pragma unroll
        for (int j = 0; j < 2; ++j) {
          const int  hin = hrow[j] + hoff;
          const int  hc  = iclamp(hin, 0, H - 1);
          const _Float16 v = A[abase[j] + (ib + hc) * W];
          ra[j][q] = (kok && (unsigned)hin < (unsigned)H) ? v : (_Float16)0.0f;
        }
      }
#pragma unroll
      for (int j = 0; j < 2; ++j) {
        const _Float16 v = Wt[woff[j] + kc];
        rb[j][q] = (kok && gok[j]) ? v : (_Float16)0.0f;
      }
    }
  };
  auto write_lds = [&](int buf) {
#pragma unroll
    for (int j = 0; j < 2; ++j) {
      *(v4h*)&As[buf][rr + 32 * j][kq] = ra[j];   // ds_store_b64
      *(v4h*)&Bs[buf][rr + 32 * j][kq] = rb[j];
    }
  };

  v8f acc0 = {};
  v8f acc1 = {};
  const int KT = (K + BK - 1) / BK;

  // prologue: tile 0 -> LDS buffer 0
  stage_regs(0);
  write_lds(0);
  __syncthreads();

  for (int kt = 0; kt < KT; ++kt) {
    const int buf = kt & 1;

    // prefetch next K-tile into registers while WMMAs run on this one
    if (kt + 1 < KT) stage_regs(kt + 1);

    const v16h av  = *(const v16h*)&As[buf][wm + (lane & 15)][khi];
    const v16h bv0 = *(const v16h*)&Bs[buf][wn + (lane & 15)][khi];
    const v16h bv1 = *(const v16h*)&Bs[buf][wn + 16 + (lane & 15)][khi];
    acc0 = __builtin_amdgcn_wmma_f32_16x16x32_f16(false, av, false, bv0,
                                                  (short)0, acc0, false, false);
    acc1 = __builtin_amdgcn_wmma_f32_16x16x32_f16(false, av, false, bv1,
                                                  (short)0, acc1, false, false);

    if (kt + 1 < KT) write_lds(buf ^ 1);
    __syncthreads();
  }

  // ---- epilogue: bias + relu + fp16 store ----
  const int mrow0 = wm + ((lane >> 4) << 3);  // lanes 0-15: rows 0..7; 16-31: 8..15
#pragma unroll
  for (int t = 0; t < 2; ++t) {
    const int gn = n_blk + wn + t * 16 + (lane & 15);
    if (gn >= N) continue;
    const float bsv = bias[gn];
    const v8f acc = t ? acc1 : acc0;
#pragma unroll
    for (int r = 0; r < 8; ++r) {
      const int m = m_blk + mrow0 + r;
      float v = acc[r] + bsv;
      if (RELU) v = fmaxf(v, 0.0f);
      const _Float16 hv = (_Float16)v;
      if (STORE == 0) {
        out[(long)m * N + gn] = hv;
      } else {
        const int b  = m / HW_out;
        const int hw = m - b * HW_out;
        const int h  = hw / W;
        const int w  = hw - h * W;
        out[((b * N + gn) * H_out + h) * W + w] = hv;
      }
    }
  }
}

// --------------------------------- host ------------------------------------

extern "C" void kernel_launch(void* const* d_in, const int* in_sizes, int n_in,
                              void* d_out, int out_size, void* d_ws, size_t ws_size,
                              hipStream_t stream) {
  (void)in_sizes; (void)n_in; (void)out_size; (void)ws_size;
  const int B = 1024;

  char*  ws  = (char*)d_ws;
  size_t cur = 0;
  auto alloc = [&](long halves) -> _Float16* {
    _Float16* p = (_Float16*)(ws + cur);
    cur = (cur + (size_t)halves * 2 + 255) & ~(size_t)255;
    return p;
  };

  auto cvt = [&](const float* src, _Float16* dst, int n) {
    cvt_f32_f16_kernel<<<dim3((n + 255) / 256), 256, 0, stream>>>(src, dst, n);
  };
  // lin: A row-major [M,K], contiguous [M,N] output
  auto gemm_lin = [&](const _Float16* A, const _Float16* Wt, const float* bias,
                      _Float16* out, int M, int N, int K, int relu) {
    dim3 grid(M / BM, (N + BN - 1) / BN);
    if (relu)
      wmma_gemm_f16<0, 1, 0, 1><<<grid, 256, 0, stream>>>(A, Wt, bias, out,
                                                          M, N, K, 1, 1, 1, 0, 1);
    else
      wmma_gemm_f16<0, 1, 0, 0><<<grid, 256, 0, stream>>>(A, Wt, bias, out,
                                                          M, N, K, 1, 1, 1, 0, 1);
  };
  // conv: im2col gather from [B,C,H,W], scatter to [B,N,H_out,W]
  auto gemm_conv = [&](const _Float16* A, const _Float16* Wt, const float* bias,
                       _Float16* out, int N, int C, int H, int W,
                       int kh, int pad, int H_out) {
    const int M = B * H_out * W;
    const int K = C * kh;
    dim3 grid(M / BM, (N + BN - 1) / BN);
    switch (kh) {
      case 1:
        wmma_gemm_f16<1, 1, 1, 0><<<grid, 256, 0, stream>>>(A, Wt, bias, out,
                                                            M, N, K, C, H, W, pad, H_out);
        break;
      case 2:
        wmma_gemm_f16<1, 2, 1, 0><<<grid, 256, 0, stream>>>(A, Wt, bias, out,
                                                            M, N, K, C, H, W, pad, H_out);
        break;
      default:
        wmma_gemm_f16<1, 3, 1, 0><<<grid, 256, 0, stream>>>(A, Wt, bias, out,
                                                            M, N, K, C, H, W, pad, H_out);
        break;
    }
  };

  // ---- inputs (setup_inputs order) ----
  const float* x   = (const float*)d_in[0];
  const float* W0a = (const float*)d_in[1];  const float* b0a = (const float*)d_in[2];
  const float* Wl0 = (const float*)d_in[3];  const float* bl0 = (const float*)d_in[4];
  const float* W0b = (const float*)d_in[5];  const float* b0b = (const float*)d_in[6];
  const float* Wr0 = (const float*)d_in[7];  const float* br0 = (const float*)d_in[8];
  const float* W0c = (const float*)d_in[9];  const float* b0c = (const float*)d_in[10];
  const float* W1  = (const float*)d_in[11]; const float* b1  = (const float*)d_in[12];
  const float* Wl1 = (const float*)d_in[13]; const float* bl1 = (const float*)d_in[14];
  const float* W2  = (const float*)d_in[15]; const float* b2  = (const float*)d_in[16];
  const float* Wra = (const float*)d_in[17]; const float* bra = (const float*)d_in[18];
  const float* Wa  = (const float*)d_in[19]; const float* ba  = (const float*)d_in[20];
  const float* W3  = (const float*)d_in[21]; const float* b3  = (const float*)d_in[22];
  const float* Wl2 = (const float*)d_in[23]; const float* bl2 = (const float*)d_in[24];
  const float* W4  = (const float*)d_in[25]; const float* b4  = (const float*)d_in[26];
  const float* Wrb = (const float*)d_in[27]; const float* brb = (const float*)d_in[28];
  const float* Wb  = (const float*)d_in[29]; const float* bb  = (const float*)d_in[30];

  // ---- fp16 weights (conv weights [O,I,kh,1] flatten to [N, K=I*kh]) ----
  _Float16* hW0a = alloc(15);     cvt(W0a, hW0a, 15);
  _Float16* hWl0 = alloc(10287);  cvt(Wl0, hWl0, 10287);
  _Float16* hW0b = alloc(375);    cvt(W0b, hW0b, 375);
  _Float16* hWr0 = alloc(10287);  cvt(Wr0, hWr0, 10287);
  _Float16* hW0c = alloc(25);     cvt(W0c, hW0c, 25);
  _Float16* hW1  = alloc(5625);   cvt(W1,  hW1,  5625);
  _Float16* hWl1 = alloc(3321);   cvt(Wl1, hWl1, 3321);
  _Float16* hW2  = alloc(33750);  cvt(W2,  hW2,  33750);
  _Float16* hWra = alloc(3321);   cvt(Wra, hWra, 3321);
  _Float16* hWa  = alloc(3750);   cvt(Wa,  hWa,  3750);
  _Float16* hW3  = alloc(90000);  cvt(W3,  hW3,  90000);
  _Float16* hWl2 = alloc(738);    cvt(Wl2, hWl2, 738);
  _Float16* hW4  = alloc(307200); cvt(W4,  hW4,  307200);
  _Float16* hWrb = alloc(738);    cvt(Wrb, hWrb, 738);
  _Float16* hWb  = alloc(230400); cvt(Wb,  hWb,  230400);

  // ---- fp16 activations ----
  _Float16* xh  = alloc((long)B * 635);           // [B,1,5,127]
  _Float16* t0  = alloc((long)B * 5 * 5 * 127);   // [B,5,5,127]
  _Float16* t1  = alloc((long)B * 5 * 5 * 81);    // [B,5,5,81]
  _Float16* t2  = alloc((long)B * 25 * 5 * 81);   // [B,25,5,81]
  _Float16* t3  = alloc((long)B * 5 * 81);        // [B,1,5,81]
  _Float16* t4  = alloc((long)B * 25 * 5 * 81);
  _Float16* t5  = alloc((long)B * 25 * 5 * 81);   // x1
  _Float16* t6  = alloc((long)B * 75 * 5 * 81);
  _Float16* t7  = alloc((long)B * 75 * 5 * 41);
  _Float16* t8  = alloc((long)B * 150 * 5 * 41);
  _Float16* t9  = alloc((long)B * 25 * 5 * 41);
  _Float16* t10 = alloc((long)B * 150 * 5 * 41);
  _Float16* t11 = alloc((long)B * 150 * 5 * 41);  // x2
  _Float16* t12 = alloc((long)B * 300 * 4 * 41);
  _Float16* t13 = alloc((long)B * 300 * 4 * 18);
  _Float16* t14 = alloc((long)B * 512 * 3 * 18);
  _Float16* t15 = alloc((long)B * 150 * 5 * 18);
  _Float16* t16 = alloc((long)B * 512 * 3 * 18);

  cvt(x, xh, B * 635);

  // ---- block 0 ----
  gemm_conv(xh, hW0a, b0a, t0, 5, 1, 5, 127, 3, 1, 5);        // [B,5,5,127]
  gemm_lin (t0, hWl0, bl0, t1, B * 5 * 5, 81, 127, 1);        // relu -> [B,5,5,81]
  gemm_conv(t1, hW0b, b0b, t2, 25, 5, 5, 81, 3, 1, 5);        // [B,25,5,81]
  gemm_lin (xh, hWr0, br0, t3, B * 5, 81, 127, 0);            // [B,1,5,81]
  gemm_conv(t3, hW0c, b0c, t4, 25, 1, 5, 81, 1, 0, 5);        // [B,25,5,81]
  {
    int n = B * 25 * 5 * 81;
    add_f16_kernel<<<dim3((n + 255) / 256), 256, 0, stream>>>(t2, t4, t5, n);
  }

  // ---- block 1 ----
  gemm_conv(t5, hW1, b1, t6, 75, 25, 5, 81, 3, 1, 5);         // [B,75,5,81]
  gemm_lin (t6, hWl1, bl1, t7, B * 75 * 5, 41, 81, 1);        // relu -> [B,75,5,41]
  gemm_conv(t7, hW2, b2, t8, 150, 75, 5, 41, 3, 1, 5);        // [B,150,5,41]
  gemm_lin (t5, hWra, bra, t9, B * 25 * 5, 41, 81, 0);        // [B,25,5,41]
  gemm_conv(t9, hWa, ba, t10, 150, 25, 5, 41, 1, 0, 5);       // [B,150,5,41]
  {
    int n = B * 150 * 5 * 41;
    add_f16_kernel<<<dim3((n + 255) / 256), 256, 0, stream>>>(t8, t10, t11, n);
  }

  // ---- block 2 ----
  gemm_conv(t11, hW3, b3, t12, 300, 150, 5, 41, 2, 0, 4);     // [B,300,4,41]
  gemm_lin (t12, hWl2, bl2, t13, B * 300 * 4, 18, 41, 1);     // relu -> [B,300,4,18]
  gemm_conv(t13, hW4, b4, t14, 512, 300, 4, 18, 2, 0, 3);     // [B,512,3,18]
  gemm_lin (t11, hWrb, brb, t15, B * 150 * 5, 18, 41, 0);     // [B,150,5,18]
  gemm_conv(t15, hWb, bb, t16, 512, 150, 5, 18, 3, 0, 3);     // [B,512,3,18]
  {
    int n = B * 512 * 3 * 18;
    add_f16_out_f32_kernel<<<dim3((n + 255) / 256), 256, 0, stream>>>(
        t14, t16, (float*)d_out, n);
  }
}
